// MultiHeadAttention_26766236189117
// MI455X (gfx1250) — matrix operand order repaired and checked by reading, not compiled, not run on hardware
//
#include <hip/hip_runtime.h>
#include <hip/hip_bf16.h>

// MI455X (gfx1250) fused multi-head attention, bf16 WMMA + fp32 accumulate.
// B=2, S=2048, D=768, H=12, Dh=64.

#define D_MODEL   768
#define NUM_HEADS 12
#define HEAD_DIM  64
#define BATCH     2
#define SEQ       2048
#define M_TOTAL   (BATCH * SEQ)   // 4096 rows for the projection GEMMs

typedef __attribute__((ext_vector_type(16))) __bf16 v16bf;
typedef __attribute__((ext_vector_type(8)))  float  v8f;

union Frag {
    v16bf        v;
    unsigned int u[8];
};

__device__ __forceinline__ unsigned short f2bf(float f) {
    unsigned int u = __float_as_uint(f);
    u += 0x7FFFu + ((u >> 16) & 1u);     // round-to-nearest-even
    return (unsigned short)(u >> 16);
}

// ---------------------------------------------------------------------------
// fp32 -> bf16 elementwise conversion (activations)
// ---------------------------------------------------------------------------
__global__ void cvt_f32_bf16_kernel(const float* __restrict__ src,
                                    unsigned short* __restrict__ dst, int n) {
    int i = blockIdx.x * blockDim.x + threadIdx.x;
    if (i < n) dst[i] = f2bf(src[i]);
}

// ---------------------------------------------------------------------------
// fp32 [K,N] -> bf16 [N,K] transpose-convert (weights; one-time cost so that
// WMMA B-fragments become contiguous 16B global loads, no LDS staging needed)
// ---------------------------------------------------------------------------
__global__ void cvt_transpose_kernel(const float* __restrict__ src,
                                     unsigned short* __restrict__ dst,
                                     int K, int N) {
    int i = blockIdx.x * blockDim.x + threadIdx.x;
    if (i < K * N) {
        const int k = i / N;
        const int n = i % N;
        dst[(size_t)n * K + k] = f2bf(src[i]);
    }
}

// ---------------------------------------------------------------------------
// GEMM: out[M,N] = A[M,K](bf16, row-major) * WT[N,K](bf16, pre-transposed)
//        + bias[N](f32).   mode 0 -> f32 output, mode 1 -> bf16 output.
// WG = 128 (4 waves). Block tile 128x64, wave tile 32x64 (2 M-tiles x 4 N-tiles),
// K-step 32. All fragments are direct 16B-aligned global b128 loads, no LDS.
// All 4 B-fragments are loaded before the wmma burst so the scheduler can use
// partial s_wait_loadcnt and overlap loads with matrix ops.
// ---------------------------------------------------------------------------
__global__ __launch_bounds__(128) void gemm_bias_kernel(
    const unsigned short* __restrict__ A,
    const unsigned short* __restrict__ WT,
    const float* __restrict__ bias,
    float* __restrict__ outF,
    unsigned short* __restrict__ outB,
    int M, int N, int K, int mode)
{
    const int t      = threadIdx.x;
    const int wave   = t >> 5;
    const int lane   = t & 31;
    const int lanelo = lane & 15;
    const int lanehi = lane >> 4;
    const int mbase  = blockIdx.y * 128 + wave * 32;
    const int nbase  = blockIdx.x * 64;

    v8f acc[2][4];
#pragma unroll
    for (int mt = 0; mt < 2; mt++)
#pragma unroll
        for (int nt = 0; nt < 4; nt++)
#pragma unroll
            for (int i = 0; i < 8; i++) acc[mt][nt][i] = 0.0f;

    for (int kb = 0; kb < K; kb += 32) {
        // A fragments (16x32 bf16, contiguous k = 16*lanehi + s, matching B): two b128 loads each.
        Frag Af[2];
#pragma unroll
        for (int mt = 0; mt < 2; mt++)
#pragma unroll
            for (int v = 0; v < 8; v++) {
                const int k0 = 2 * v + 16 * lanehi;
                Af[mt].u[v] = *(const unsigned int*)
                    &A[(size_t)(mbase + mt * 16 + lanelo) * K + kb + k0];
            }

        // All four B fragments up front (8x global_load_b128 clause).
        Frag Bf[4];
#pragma unroll
        for (int nt = 0; nt < 4; nt++)
#pragma unroll
            for (int v = 0; v < 8; v++)
                Bf[nt].u[v] = *(const unsigned int*)
                    &WT[(size_t)(nbase + nt * 16 + lanelo) * K + kb + 2 * v + 16 * lanehi];

        if (kb + 32 < K) {
            __builtin_prefetch(&A[(size_t)(mbase + lanelo) * K + kb + 32], 0, 1);
            __builtin_prefetch(&WT[(size_t)(nbase + lanelo) * K + kb + 32], 0, 1);
        }

#pragma unroll
        for (int nt = 0; nt < 4; nt++)
#pragma unroll
            for (int mt = 0; mt < 2; mt++)
                acc[mt][nt] = __builtin_amdgcn_wmma_f32_16x16x32_bf16(
                    false, Af[mt].v, false, Bf[nt].v, (short)0, acc[mt][nt], false, false);
    }

    // Epilogue: bias + store. C layout: row = v + 8*lanehi, col = lanelo.
#pragma unroll
    for (int nt = 0; nt < 4; nt++) {
        const int n    = nbase + nt * 16 + lanelo;
        const float bv = bias[n];
#pragma unroll
        for (int mt = 0; mt < 2; mt++)
#pragma unroll
            for (int v = 0; v < 8; v++) {
                const int m   = mbase + mt * 16 + v + 8 * lanehi;
                const float r = acc[mt][nt][v] + bv;
                if (mode == 0) outF[(size_t)m * N + n] = r;
                else           outB[(size_t)m * N + n] = f2bf(r);
            }
    }
}

// ---------------------------------------------------------------------------
// Fused flash attention (no causal mask): per (b,h) and 64-query block.
// Q,K,V are bf16 in [B,S,H*Dh] layout. Output ctx bf16 in same layout.
// WG = 128 (4 waves); each wave owns 16 query rows x full Dh=64.
// ---------------------------------------------------------------------------
__global__ __launch_bounds__(128) void attn_kernel(
    const unsigned short* __restrict__ Q,
    const unsigned short* __restrict__ Km,
    const unsigned short* __restrict__ V,
    unsigned short* __restrict__ Ctx)
{
    __shared__ unsigned short VT[64 * 64];        // [dh][key] transposed V block
    __shared__ unsigned short PT[4 * 16 * 64];    // per-wave P tile (16 q x 64 key)

    const int t      = threadIdx.x;
    const int wave   = t >> 5;
    const int lane   = t & 31;
    const int lanelo = lane & 15;
    const int lanehi = lane >> 4;
    const int b      = blockIdx.y / NUM_HEADS;
    const int h      = blockIdx.y % NUM_HEADS;
    const int qbase  = blockIdx.x * 64 + wave * 16;
    const float scale = 0.125f;  // 1/sqrt(64)

    auto rowoff = [&](int s) -> size_t {
        return ((size_t)(b * SEQ + s)) * D_MODEL + h * HEAD_DIM;
    };

    // Q fragments: two 16x32 bf16 A-fragments covering Dh=64.
    Frag Qf[2];
#pragma unroll
    for (int f = 0; f < 2; f++)
#pragma unroll
        for (int v = 0; v < 8; v++) {
            const int k0 = 2 * v + 16 * lanehi + 32 * f;
            Qf[f].u[v] = *(const unsigned int*)&Q[rowoff(qbase + lanelo) + k0];
        }

    float m_run[8], l_run[8];
    v8f   O[4];
#pragma unroll
    for (int v = 0; v < 8; v++) { m_run[v] = -1e30f; l_run[v] = 0.0f; }
#pragma unroll
    for (int a = 0; a < 4; a++)
#pragma unroll
        for (int i = 0; i < 8; i++) O[a][i] = 0.0f;

    for (int kb = 0; kb < SEQ; kb += 64) {
        // Stage V[kb..kb+64, :] transposed into VT[dh][key].
        {
            const int keyloc = t & 63;
            const int dhhalf = (t >> 6) * 32;
#pragma unroll
            for (int j = 0; j < 16; j++) {
                unsigned int u = *(const unsigned int*)&V[rowoff(kb + keyloc) + dhhalf + 2 * j];
                VT[(dhhalf + 2 * j)     * 64 + keyloc] = (unsigned short)(u & 0xFFFFu);
                VT[(dhhalf + 2 * j + 1) * 64 + keyloc] = (unsigned short)(u >> 16);
            }
        }
        __syncthreads();

        // Scores S[16 q x 64 key] = Q * K^T, fp32 accumulate over Dh=64.
        // Load all four K B-fragments per Dh-chunk before the wmma burst.
        v8f S[4];
#pragma unroll
        for (int nt = 0; nt < 4; nt++)
#pragma unroll
            for (int i = 0; i < 8; i++) S[nt][i] = 0.0f;

#pragma unroll
        for (int f = 0; f < 2; f++) {
            Frag Kf[4];   // B-fragments of K^T: kk = dh, n = key
#pragma unroll
            for (int nt = 0; nt < 4; nt++)
#pragma unroll
                for (int v = 0; v < 8; v++)
                    Kf[nt].u[v] = *(const unsigned int*)&Km[rowoff(kb + nt * 16 + lanelo)
                                                           + 32 * f + 2 * v + 16 * lanehi];
#pragma unroll
            for (int nt = 0; nt < 4; nt++)
                S[nt] = __builtin_amdgcn_wmma_f32_16x16x32_bf16(
                    false, Qf[f].v, false, Kf[nt].v, (short)0, S[nt], false, false);
        }

        // Online softmax. Row m = v + 8*lanehi lives across lanelo + 4 tiles.
        float corr[8];
#pragma unroll
        for (int v = 0; v < 8; v++) {
            float x = fmaxf(fmaxf(S[0][v], S[1][v]), fmaxf(S[2][v], S[3][v]));
            x = fmaxf(x, __shfl_xor(x, 1));
            x = fmaxf(x, __shfl_xor(x, 2));
            x = fmaxf(x, __shfl_xor(x, 4));
            x = fmaxf(x, __shfl_xor(x, 8));
            x *= scale;                          // scale > 0 commutes with max
            const float mnew = fmaxf(m_run[v], x);
            corr[v] = __expf(m_run[v] - mnew);
            m_run[v] = mnew;
        }

#pragma unroll
        for (int v = 0; v < 8; v++) {
            float rs = 0.0f;
#pragma unroll
            for (int a = 0; a < 4; a++) {
                const float p = __expf(fmaf(S[a][v], scale, -m_run[v]));
                rs += p;
                PT[wave * 1024 + (v + 8 * lanehi) * 64 + a * 16 + lanelo] = f2bf(p);
            }
            rs += __shfl_xor(rs, 1);
            rs += __shfl_xor(rs, 2);
            rs += __shfl_xor(rs, 4);
            rs += __shfl_xor(rs, 8);
            l_run[v] = l_run[v] * corr[v] + rs;
#pragma unroll
            for (int a = 0; a < 4; a++) O[a][v] *= corr[v];
        }
        __syncthreads();   // PT visible (re-layout C->A goes through LDS)

        // O += P * V  (A = P 16x64, B = V 64key x 16dh per tile)
#pragma unroll
        for (int f = 0; f < 2; f++) {
            Frag Pf;
#pragma unroll
            for (int v = 0; v < 8; v++) {
                const int k0 = 2 * v + 16 * lanehi + 32 * f;
                Pf.u[v] = *(const unsigned int*)&PT[wave * 1024 + lanelo * 64 + k0];
            }
            Frag Vf[4];   // B-fragments: kk = key (pairs contiguous in VT rows), n = dh
#pragma unroll
            for (int a = 0; a < 4; a++)
#pragma unroll
                for (int v = 0; v < 8; v++)
                    Vf[a].u[v] = *(const unsigned int*)&VT[(a * 16 + lanelo) * 64
                                                           + 32 * f + 2 * v + 16 * lanehi];
#pragma unroll
            for (int a = 0; a < 4; a++)
                O[a] = __builtin_amdgcn_wmma_f32_16x16x32_bf16(
                    false, Pf.v, false, Vf[a].v, (short)0, O[a], false, false);
        }
        __syncthreads();   // protect VT before next iteration restages it
    }

    // Normalize and store ctx (bf16, [B,S,H*Dh] layout == reference merge order).
#pragma unroll
    for (int a = 0; a < 4; a++)
#pragma unroll
        for (int v = 0; v < 8; v++) {
            const float r = O[a][v] / l_run[v];
            Ctx[rowoff(qbase + v + 8 * lanehi) + a * 16 + lanelo] = f2bf(r);
        }
}

// ---------------------------------------------------------------------------
// Host launcher
// ---------------------------------------------------------------------------
extern "C" void kernel_launch(void* const* d_in, const int* in_sizes, int n_in,
                              void* d_out, int out_size, void* d_ws, size_t ws_size,
                              hipStream_t stream) {
    (void)in_sizes; (void)n_in; (void)out_size; (void)ws_size;

    const float* x  = (const float*)d_in[0];
    const float* Wq = (const float*)d_in[1];
    const float* bq = (const float*)d_in[2];
    const float* Wk = (const float*)d_in[3];
    const float* bk = (const float*)d_in[4];
    const float* Wv = (const float*)d_in[5];
    const float* bv = (const float*)d_in[6];
    const float* Wo = (const float*)d_in[7];
    const float* bo = (const float*)d_in[8];

    char*  ws  = (char*)d_ws;
    size_t off = 0;
    auto carve = [&](size_t bytes) -> void* {
        void* p = ws + off;
        off = (off + bytes + 255) & ~(size_t)255;
        return p;
    };

    const size_t actBytes = (size_t)M_TOTAL * D_MODEL * sizeof(unsigned short);
    const size_t wBytes   = (size_t)D_MODEL * D_MODEL * sizeof(unsigned short);

    unsigned short* xb   = (unsigned short*)carve(actBytes);
    unsigned short* wqT  = (unsigned short*)carve(wBytes);   // [N,K] bf16
    unsigned short* wkT  = (unsigned short*)carve(wBytes);
    unsigned short* wvT  = (unsigned short*)carve(wBytes);
    unsigned short* woT  = (unsigned short*)carve(wBytes);
    unsigned short* qb   = (unsigned short*)carve(actBytes);
    unsigned short* kb   = (unsigned short*)carve(actBytes);
    unsigned short* vb   = (unsigned short*)carve(actBytes);
    unsigned short* ctxb = (unsigned short*)carve(actBytes);

    // 1) fp32 -> bf16 conversions (x straight; weights transposed to [N,K])
    const int nAct = M_TOTAL * D_MODEL;
    const int nW   = D_MODEL * D_MODEL;
    cvt_f32_bf16_kernel<<<(nAct + 255) / 256, 256, 0, stream>>>(x, xb, nAct);
    cvt_transpose_kernel<<<(nW + 255) / 256, 256, 0, stream>>>(Wq, wqT, D_MODEL, D_MODEL);
    cvt_transpose_kernel<<<(nW + 255) / 256, 256, 0, stream>>>(Wk, wkT, D_MODEL, D_MODEL);
    cvt_transpose_kernel<<<(nW + 255) / 256, 256, 0, stream>>>(Wv, wvT, D_MODEL, D_MODEL);
    cvt_transpose_kernel<<<(nW + 255) / 256, 256, 0, stream>>>(Wo, woT, D_MODEL, D_MODEL);

    // 2) Q/K/V projections (bf16 out)
    dim3 gg(D_MODEL / 64, M_TOTAL / 128);
    gemm_bias_kernel<<<gg, 128, 0, stream>>>(xb, wqT, bq, nullptr, qb,
                                             M_TOTAL, D_MODEL, D_MODEL, 1);
    gemm_bias_kernel<<<gg, 128, 0, stream>>>(xb, wkT, bk, nullptr, kb,
                                             M_TOTAL, D_MODEL, D_MODEL, 1);
    gemm_bias_kernel<<<gg, 128, 0, stream>>>(xb, wvT, bv, nullptr, vb,
                                             M_TOTAL, D_MODEL, D_MODEL, 1);

    // 3) fused flash attention
    dim3 ga(SEQ / 64, BATCH * NUM_HEADS);
    attn_kernel<<<ga, 128, 0, stream>>>(qb, kb, vb, ctxb);

    // 4) output projection (fp32 out)
    gemm_bias_kernel<<<gg, 128, 0, stream>>>(ctxb, woT, bo, (float*)d_out, nullptr,
                                             M_TOTAL, D_MODEL, D_MODEL, 0);
}
